// LongformerSelfAttentionForT5_80573586473121
// MI455X (gfx1250) — compile-verified, matrix operand order checked by reading, not run on hardware
//
#include <hip/hip_runtime.h>

#define B_  2
#define S_  4096
#define D_  768
#define H_  12
#define DH_ 64
#define WC_ 256
#define NC_ (S_ / WC_)
#define G_  64
#define NEGV (-1000000000.0f)

typedef __attribute__((ext_vector_type(16))) __bf16        v16bf;
typedef __attribute__((ext_vector_type(8)))  float         v8f;
typedef __attribute__((ext_vector_type(4)))  unsigned int  u32x4;

union Frag {
    u32x4 u[2];
    v16bf v;
    unsigned short s[16];
};

union CF {
    v8f v;
    float f[8];
};

static __device__ __forceinline__ unsigned short f2bf(float f) {
    unsigned int u = __float_as_uint(f);
    u += 0x7FFFu + ((u >> 16) & 1u);
    return (unsigned short)(u >> 16);
}

static __device__ __forceinline__ v8f wmma_bf16(v16bf a, v16bf b, v8f c) {
    return __builtin_amdgcn_wmma_f32_16x16x32_bf16(false, a, false, b, (short)0, c,
                                                   false, false);
}

// ---------------------------------------------------------------------------
// fp32 -> bf16 conversion
// ---------------------------------------------------------------------------
__global__ void f32_to_bf16_kernel(const float* __restrict__ in,
                                   unsigned short* __restrict__ out, int n) {
    int i = blockIdx.x * blockDim.x + threadIdx.x;
    if (i < n) out[i] = f2bf(in[i]);
}

// ---------------------------------------------------------------------------
// WMMA GEMM: C[M,N] = A[M,K] (bf16) @ B[K,N] (bf16)
// 128x64 tile per workgroup; 8 waves, each 32x32 (2x2 WMMA tiles).
// MODE 0: store bf16 with head remap [B,S,H*DH] -> [B,H,S,DH]
// MODE 1: store fp32 plain row-major
// ---------------------------------------------------------------------------
template <int MODE>
__global__ __launch_bounds__(256) void gemm_bf16_wmma(
    const unsigned short* __restrict__ A, const unsigned short* __restrict__ Bw,
    unsigned short* __restrict__ Obf, float* __restrict__ Of, int M, int N, int K) {
    __shared__ unsigned short lA[128][40];
    __shared__ unsigned short lB[64][40];  // transposed: [n][k]

    const int m0 = blockIdx.x * 128;
    const int n0 = blockIdx.y * 64;
    const int tid  = threadIdx.x;
    const int lane = tid & 31;
    const int wave = tid >> 5;
    const int wm   = wave >> 1;   // 0..3 -> 32-row group
    const int wn   = wave & 1;    // 0..1 -> 32-col group
    const int half = lane >> 4;
    const int l16  = lane & 15;

    CF acc[2][2];
    for (int rt = 0; rt < 2; ++rt)
        for (int ct = 0; ct < 2; ++ct)
            for (int r = 0; r < 8; ++r) acc[rt][ct].f[r] = 0.f;

    for (int kb = 0; kb < K; kb += 32) {
        {   // A tile 128x32 bf16, 32 B per thread
            int row = tid >> 1;
            int cb  = (tid & 1) * 16;
            const unsigned short* p = A + (size_t)(m0 + row) * K + kb + cb;
            u32x4 d0 = *(const u32x4*)p;
            u32x4 d1 = *(const u32x4*)(p + 8);
            *(u32x4*)&lA[row][cb]     = d0;
            *(u32x4*)&lA[row][cb + 8] = d1;
        }
        {   // B tile 32x64, transposed into lB[n][k]
            int kk = tid >> 3;
            int nb = (tid & 7) * 8;
            const unsigned short* p = Bw + (size_t)(kb + kk) * N + n0 + nb;
            union { u32x4 u; unsigned short s[8]; } w;
            w.u = *(const u32x4*)p;
            for (int e = 0; e < 8; ++e) lB[nb + e][kk] = w.s[e];
        }
        __syncthreads();

        Frag fa[2], fb[2];
        for (int rt = 0; rt < 2; ++rt) {
            int row = wm * 32 + rt * 16 + l16;
            fa[rt].u[0] = *(const u32x4*)&lA[row][8 * half];
            fa[rt].u[1] = *(const u32x4*)&lA[row][16 + 8 * half];
        }
        for (int ct = 0; ct < 2; ++ct) {
            int col = wn * 32 + ct * 16 + l16;
            fb[ct].u[0] = *(const u32x4*)&lB[col][8 * half];
            fb[ct].u[1] = *(const u32x4*)&lB[col][16 + 8 * half];
        }
        for (int rt = 0; rt < 2; ++rt)
            for (int ct = 0; ct < 2; ++ct)
                acc[rt][ct].v = wmma_bf16(fa[rt].v, fb[ct].v, acc[rt][ct].v);
        __syncthreads();
    }

    for (int rt = 0; rt < 2; ++rt)
        for (int ct = 0; ct < 2; ++ct)
            for (int r = 0; r < 8; ++r) {
                int gr = m0 + wm * 32 + rt * 16 + r + 8 * half;
                int gc = n0 + wn * 32 + ct * 16 + l16;
                float val = acc[rt][ct].f[r];
                if (MODE == 0) {
                    int bb = gr / S_, s = gr % S_;
                    int hh = gc / DH_, dh = gc % DH_;
                    size_t dst = (((size_t)(bb * H_ + hh)) * S_ + s) * DH_ + dh;
                    Obf[dst] = f2bf(val);
                } else {
                    Of[(size_t)gr * N + gc] = val;
                }
            }
}

// ---------------------------------------------------------------------------
// Banded attention (flash style). One workgroup per (b, h, chunk t).
// 8 waves x 32 rows. 13 column blocks of 64: blk 0 = global keys 0..63,
// blk 1..12 = band keys j = t*c + s - c, s in [(blk-1)*64, blk*64).
// ---------------------------------------------------------------------------
__global__ __launch_bounds__(256) void band_attn_kernel(
    const unsigned short* __restrict__ q, const unsigned short* __restrict__ k,
    const unsigned short* __restrict__ v, unsigned short* __restrict__ attn) {
    __shared__ unsigned short vt[64][72];        // transposed V block [dh][key]
    __shared__ unsigned short pls[8][32][72];    // per-wave P tile [row][col]

    const int bi = blockIdx.x;
    const int t  = bi % NC_;
    const int h  = (bi / NC_) % H_;
    const int b  = bi / (NC_ * H_);

    const unsigned short* qh = q + ((size_t)(b * H_ + h) * S_) * DH_;
    const unsigned short* kh = k + ((size_t)(b * H_ + h) * S_) * DH_;
    const unsigned short* vh = v + ((size_t)(b * H_ + h) * S_) * DH_;

    const int tid  = threadIdx.x;
    const int lane = tid & 31;
    const int wave = tid >> 5;
    const int half = lane >> 4;
    const int l16  = lane & 15;
    const int rbase = wave * 32;

    // Q fragments (reused across all column blocks)
    Frag aq[2][2];
    for (int rt = 0; rt < 2; ++rt)
        for (int ks = 0; ks < 2; ++ks) {
            int row = t * WC_ + rbase + rt * 16 + l16;
            const unsigned short* p = qh + (size_t)row * DH_ + ks * 32;
            aq[rt][ks].u[0] = *(const u32x4*)(p + 8 * half);
            aq[rt][ks].u[1] = *(const u32x4*)(p + 16 + 8 * half);
        }

    CF acc[2][4];
    float mrow[2][8], lrow[2][8];
    for (int rt = 0; rt < 2; ++rt)
        for (int r = 0; r < 8; ++r) {
            mrow[rt][r] = -3.0e38f;
            lrow[rt][r] = 0.f;
            for (int dt = 0; dt < 4; ++dt) acc[rt][dt].f[r] = 0.f;
        }

    for (int blk = 0; blk < 13; ++blk) {
        // ---- stage V block (64 keys x 64 dh) transposed into LDS ----
        {
            int i   = tid >> 2;          // key index in block 0..63
            int dhb = (tid & 3) * 16;
            int key; bool valid;
            if (blk == 0) { key = i; valid = true; }
            else {
                int s = (blk - 1) * 64 + i;
                int j = t * WC_ + s - WC_;
                key = j; valid = (j >= 0) && (j < S_);
            }
            Frag fv;
            if (valid) {
                const unsigned short* p = vh + (size_t)key * DH_ + dhb;
                fv.u[0] = *(const u32x4*)p;
                fv.u[1] = *(const u32x4*)(p + 8);
            } else {
                fv.u[0] = (u32x4){0u, 0u, 0u, 0u};
                fv.u[1] = (u32x4){0u, 0u, 0u, 0u};
            }
            for (int e = 0; e < 16; ++e) vt[dhb + e][i] = fv.s[e];
        }
        __syncthreads();

        // ---- K fragments ----
        Frag bk[4][2];
        for (int ct = 0; ct < 4; ++ct) {
            int i = ct * 16 + l16;
            int key; bool valid;
            if (blk == 0) { key = i; valid = true; }
            else {
                int s = (blk - 1) * 64 + i;
                int j = t * WC_ + s - WC_;
                key = j; valid = (j >= 0) && (j < S_);
            }
            for (int ks = 0; ks < 2; ++ks) {
                if (valid) {
                    const unsigned short* p = kh + (size_t)key * DH_ + ks * 32;
                    bk[ct][ks].u[0] = *(const u32x4*)(p + 8 * half);
                    bk[ct][ks].u[1] = *(const u32x4*)(p + 16 + 8 * half);
                } else {
                    bk[ct][ks].u[0] = (u32x4){0u, 0u, 0u, 0u};
                    bk[ct][ks].u[1] = (u32x4){0u, 0u, 0u, 0u};
                }
            }
        }

        // ---- scores = Q @ K^T ----
        CF sc[2][4];
        for (int rt = 0; rt < 2; ++rt)
            for (int ct = 0; ct < 4; ++ct) {
                CF z; for (int r = 0; r < 8; ++r) z.f[r] = 0.f;
                z.v = wmma_bf16(aq[rt][0].v, bk[ct][0].v, z.v);
                z.v = wmma_bf16(aq[rt][1].v, bk[ct][1].v, z.v);
                sc[rt][ct] = z;
            }

        // ---- mask + online softmax ----
        for (int rt = 0; rt < 2; ++rt) {
            float nm[8];
            for (int r = 0; r < 8; ++r) nm[r] = mrow[rt][r];
            for (int ct = 0; ct < 4; ++ct) {
                int i = ct * 16 + l16;
                for (int r = 0; r < 8; ++r) {
                    int rrow = rbase + rt * 16 + r + 8 * half;  // row in chunk
                    float x = sc[rt][ct].f[r];
                    bool ok;
                    if (blk == 0) ok = true;
                    else {
                        int s = (blk - 1) * 64 + i;
                        int j = t * WC_ + s - WC_;
                        ok = (s >= rrow) && (s <= rrow + 2 * WC_) &&
                             (j >= 0) && (j < S_);
                    }
                    if (!ok) x = NEGV;
                    sc[rt][ct].f[r] = x;
                    nm[r] = fmaxf(nm[r], x);
                }
            }
            for (int r = 0; r < 8; ++r)
                for (int off = 1; off < 16; off <<= 1)
                    nm[r] = fmaxf(nm[r], __shfl_xor(nm[r], off, 32));

            float esc[8], ps[8];
            for (int r = 0; r < 8; ++r) {
                esc[r] = __expf(mrow[rt][r] - nm[r]);
                mrow[rt][r] = nm[r];
                ps[r] = 0.f;
            }
            for (int ct = 0; ct < 4; ++ct)
                for (int r = 0; r < 8; ++r) {
                    float p = __expf(sc[rt][ct].f[r] - nm[r]);
                    sc[rt][ct].f[r] = p;
                    ps[r] += p;
                }
            for (int r = 0; r < 8; ++r) {
                for (int off = 1; off < 16; off <<= 1)
                    ps[r] += __shfl_xor(ps[r], off, 32);
                lrow[rt][r] = lrow[rt][r] * esc[r] + ps[r];
            }
            for (int dt = 0; dt < 4; ++dt)
                for (int r = 0; r < 8; ++r) acc[rt][dt].f[r] *= esc[r];

            // write P tile (C-layout -> LDS) for A-fragment reload
            for (int ct = 0; ct < 4; ++ct)
                for (int r = 0; r < 8; ++r)
                    pls[wave][rt * 16 + r + 8 * half][ct * 16 + l16] =
                        f2bf(sc[rt][ct].f[r]);
        }

        // ---- P @ V accumulate ----
        Frag ap[2][2];
        for (int rt = 0; rt < 2; ++rt)
            for (int ks = 0; ks < 2; ++ks) {
                const unsigned short* p = &pls[wave][rt * 16 + l16][ks * 32];
                ap[rt][ks].u[0] = *(const u32x4*)(p + 8 * half);
                ap[rt][ks].u[1] = *(const u32x4*)(p + 16 + 8 * half);
            }
        Frag bv[4][2];
        for (int dt = 0; dt < 4; ++dt)
            for (int ks = 0; ks < 2; ++ks) {
                const unsigned short* p = &vt[dt * 16 + l16][ks * 32];
                bv[dt][ks].u[0] = *(const u32x4*)(p + 8 * half);
                bv[dt][ks].u[1] = *(const u32x4*)(p + 16 + 8 * half);
            }
        for (int rt = 0; rt < 2; ++rt)
            for (int dt = 0; dt < 4; ++dt) {
                acc[rt][dt].v = wmma_bf16(ap[rt][0].v, bv[dt][0].v, acc[rt][dt].v);
                acc[rt][dt].v = wmma_bf16(ap[rt][1].v, bv[dt][1].v, acc[rt][dt].v);
            }
        __syncthreads();
    }

    // ---- normalize + store (token-major bf16 [B,S,H*DH]) ----
    for (int rt = 0; rt < 2; ++rt)
        for (int dt = 0; dt < 4; ++dt)
            for (int r = 0; r < 8; ++r) {
                int srow = t * WC_ + rbase + rt * 16 + r + 8 * half;
                float val = acc[rt][dt].f[r] / lrow[rt][r];
                size_t dst = ((size_t)(b * S_ + srow)) * D_ + h * DH_ + dt * 16 + l16;
                attn[dst] = f2bf(val);
            }
}

// ---------------------------------------------------------------------------
// Global-token attention: 64 rows per (b,h) attend to all S keys.
// Waves 0-1 compute (32 rows each); all 8 waves help stage V blocks.
// Overwrites the first 64 tokens written by the band kernel.
// ---------------------------------------------------------------------------
__global__ __launch_bounds__(256) void global_attn_kernel(
    const unsigned short* __restrict__ qg, const unsigned short* __restrict__ kg,
    const unsigned short* __restrict__ vg, unsigned short* __restrict__ attn) {
    __shared__ unsigned short vt[64][72];
    __shared__ unsigned short pls[2][32][72];

    const int bi = blockIdx.x;
    const int h  = bi % H_;
    const int b  = bi / H_;

    const unsigned short* qh = qg + ((size_t)(b * H_ + h) * S_) * DH_;
    const unsigned short* kh = kg + ((size_t)(b * H_ + h) * S_) * DH_;
    const unsigned short* vh = vg + ((size_t)(b * H_ + h) * S_) * DH_;

    const int tid  = threadIdx.x;
    const int lane = tid & 31;
    const int wave = tid >> 5;
    const int half = lane >> 4;
    const int l16  = lane & 15;
    const bool compute = (wave < 2);

    Frag aq[2][2];
    CF acc[2][4];
    float mrow[2][8], lrow[2][8];
    for (int rt = 0; rt < 2; ++rt)
        for (int r = 0; r < 8; ++r) {
            mrow[rt][r] = -3.0e38f;
            lrow[rt][r] = 0.f;
            for (int dt = 0; dt < 4; ++dt) acc[rt][dt].f[r] = 0.f;
        }
    if (compute) {
        for (int rt = 0; rt < 2; ++rt)
            for (int ks = 0; ks < 2; ++ks) {
                int row = wave * 32 + rt * 16 + l16;  // < 64
                const unsigned short* p = qh + (size_t)row * DH_ + ks * 32;
                aq[rt][ks].u[0] = *(const u32x4*)(p + 8 * half);
                aq[rt][ks].u[1] = *(const u32x4*)(p + 16 + 8 * half);
            }
    }

    for (int kb = 0; kb < S_; kb += 64) {
        {   // stage V block transposed
            int i   = tid >> 2;
            int dhb = (tid & 3) * 16;
            const unsigned short* p = vh + (size_t)(kb + i) * DH_ + dhb;
            Frag fv;
            fv.u[0] = *(const u32x4*)p;
            fv.u[1] = *(const u32x4*)(p + 8);
            for (int e = 0; e < 16; ++e) vt[dhb + e][i] = fv.s[e];
        }
        __syncthreads();

        if (compute) {
            Frag bk[4][2];
            for (int ct = 0; ct < 4; ++ct) {
                int key = kb + ct * 16 + l16;
                for (int ks = 0; ks < 2; ++ks) {
                    const unsigned short* p = kh + (size_t)key * DH_ + ks * 32;
                    bk[ct][ks].u[0] = *(const u32x4*)(p + 8 * half);
                    bk[ct][ks].u[1] = *(const u32x4*)(p + 16 + 8 * half);
                }
            }
            CF sc[2][4];
            for (int rt = 0; rt < 2; ++rt)
                for (int ct = 0; ct < 4; ++ct) {
                    CF z; for (int r = 0; r < 8; ++r) z.f[r] = 0.f;
                    z.v = wmma_bf16(aq[rt][0].v, bk[ct][0].v, z.v);
                    z.v = wmma_bf16(aq[rt][1].v, bk[ct][1].v, z.v);
                    sc[rt][ct] = z;
                }
            for (int rt = 0; rt < 2; ++rt) {
                float nm[8];
                for (int r = 0; r < 8; ++r) nm[r] = mrow[rt][r];
                for (int ct = 0; ct < 4; ++ct)
                    for (int r = 0; r < 8; ++r)
                        nm[r] = fmaxf(nm[r], sc[rt][ct].f[r]);
                for (int r = 0; r < 8; ++r)
                    for (int off = 1; off < 16; off <<= 1)
                        nm[r] = fmaxf(nm[r], __shfl_xor(nm[r], off, 32));
                float esc[8], ps[8];
                for (int r = 0; r < 8; ++r) {
                    esc[r] = __expf(mrow[rt][r] - nm[r]);
                    mrow[rt][r] = nm[r];
                    ps[r] = 0.f;
                }
                for (int ct = 0; ct < 4; ++ct)
                    for (int r = 0; r < 8; ++r) {
                        float p = __expf(sc[rt][ct].f[r] - nm[r]);
                        sc[rt][ct].f[r] = p;
                        ps[r] += p;
                    }
                for (int r = 0; r < 8; ++r) {
                    for (int off = 1; off < 16; off <<= 1)
                        ps[r] += __shfl_xor(ps[r], off, 32);
                    lrow[rt][r] = lrow[rt][r] * esc[r] + ps[r];
                }
                for (int dt = 0; dt < 4; ++dt)
                    for (int r = 0; r < 8; ++r) acc[rt][dt].f[r] *= esc[r];
                for (int ct = 0; ct < 4; ++ct)
                    for (int r = 0; r < 8; ++r)
                        pls[wave][rt * 16 + r + 8 * half][ct * 16 + l16] =
                            f2bf(sc[rt][ct].f[r]);
            }
            Frag ap[2][2];
            for (int rt = 0; rt < 2; ++rt)
                for (int ks = 0; ks < 2; ++ks) {
                    const unsigned short* p = &pls[wave][rt * 16 + l16][ks * 32];
                    ap[rt][ks].u[0] = *(const u32x4*)(p + 8 * half);
                    ap[rt][ks].u[1] = *(const u32x4*)(p + 16 + 8 * half);
                }
            Frag bv[4][2];
            for (int dt = 0; dt < 4; ++dt)
                for (int ks = 0; ks < 2; ++ks) {
                    const unsigned short* p = &vt[dt * 16 + l16][ks * 32];
                    bv[dt][ks].u[0] = *(const u32x4*)(p + 8 * half);
                    bv[dt][ks].u[1] = *(const u32x4*)(p + 16 + 8 * half);
                }
            for (int rt = 0; rt < 2; ++rt)
                for (int dt = 0; dt < 4; ++dt) {
                    acc[rt][dt].v = wmma_bf16(ap[rt][0].v, bv[dt][0].v, acc[rt][dt].v);
                    acc[rt][dt].v = wmma_bf16(ap[rt][1].v, bv[dt][1].v, acc[rt][dt].v);
                }
        }
        __syncthreads();
    }

    if (compute) {
        for (int rt = 0; rt < 2; ++rt)
            for (int dt = 0; dt < 4; ++dt)
                for (int r = 0; r < 8; ++r) {
                    int srow = wave * 32 + rt * 16 + r + 8 * half;  // < 64
                    float val = acc[rt][dt].f[r] / lrow[rt][r];
                    size_t dst = ((size_t)(b * S_ + srow)) * D_ + h * DH_ +
                                 dt * 16 + l16;
                    attn[dst] = f2bf(val);
                }
    }
}

// ---------------------------------------------------------------------------
extern "C" void kernel_launch(void* const* d_in, const int* in_sizes, int n_in,
                              void* d_out, int out_size, void* d_ws, size_t ws_size,
                              hipStream_t stream) {
    (void)in_sizes; (void)n_in; (void)out_size; (void)ws_size;

    const float* x = (const float*)d_in[0];
    const float* Ws[7] = {(const float*)d_in[1], (const float*)d_in[2],
                          (const float*)d_in[3], (const float*)d_in[4],
                          (const float*)d_in[5], (const float*)d_in[6],
                          (const float*)d_in[7]};
    // d_in[8] = n_global (constant 64, baked in)

    unsigned char* ws = (unsigned char*)d_ws;
    size_t off = 0;
    auto alloc = [&](size_t bytes) -> void* {
        void* p = ws + off;
        off = (off + bytes + 255) & ~(size_t)255;
        return p;
    };

    const size_t NX = (size_t)B_ * S_ * D_;  // 6,291,456
    const size_t NW = (size_t)D_ * D_;       //   589,824

    unsigned short* xb = (unsigned short*)alloc(NX * 2);
    unsigned short* wb[7];
    for (int i = 0; i < 7; ++i) wb[i] = (unsigned short*)alloc(NW * 2);
    unsigned short* qb   = (unsigned short*)alloc(NX * 2);
    unsigned short* kb   = (unsigned short*)alloc(NX * 2);
    unsigned short* vb   = (unsigned short*)alloc(NX * 2);
    unsigned short* qgb  = (unsigned short*)alloc(NX * 2);
    unsigned short* kgb  = (unsigned short*)alloc(NX * 2);
    unsigned short* vgb  = (unsigned short*)alloc(NX * 2);
    unsigned short* attn = (unsigned short*)alloc(NX * 2);

    // 1) fp32 -> bf16
    f32_to_bf16_kernel<<<(int)((NX + 255) / 256), 256, 0, stream>>>(x, xb, (int)NX);
    for (int i = 0; i < 7; ++i)
        f32_to_bf16_kernel<<<(int)((NW + 255) / 256), 256, 0, stream>>>(
            Ws[i], wb[i], (int)NW);

    // 2) projections (bf16, head-major output)
    dim3 gg((B_ * S_) / 128, D_ / 64);
    gemm_bf16_wmma<0><<<gg, 256, 0, stream>>>(xb, wb[0], qb,  nullptr, B_ * S_, D_, D_);
    gemm_bf16_wmma<0><<<gg, 256, 0, stream>>>(xb, wb[1], kb,  nullptr, B_ * S_, D_, D_);
    gemm_bf16_wmma<0><<<gg, 256, 0, stream>>>(xb, wb[2], vb,  nullptr, B_ * S_, D_, D_);
    gemm_bf16_wmma<0><<<gg, 256, 0, stream>>>(xb, wb[3], qgb, nullptr, B_ * S_, D_, D_);
    gemm_bf16_wmma<0><<<gg, 256, 0, stream>>>(xb, wb[4], kgb, nullptr, B_ * S_, D_, D_);
    gemm_bf16_wmma<0><<<gg, 256, 0, stream>>>(xb, wb[5], vgb, nullptr, B_ * S_, D_, D_);

    // 3) banded attention, then global-token override (same stream => ordered)
    band_attn_kernel<<<B_ * H_ * NC_, 256, 0, stream>>>(qb, kb, vb, attn);
    global_attn_kernel<<<B_ * H_, 256, 0, stream>>>(qgb, kgb, vgb, attn);

    // 4) output projection (fp32 to d_out)
    gemm_bf16_wmma<1><<<gg, 256, 0, stream>>>(attn, wb[6], nullptr, (float*)d_out,
                                              B_ * S_, D_, D_);
}
